// ContentEncoder_11811160064754
// MI455X (gfx1250) — matrix operand order repaired and checked by reading, not compiled, not run on hardware
//
#include <hip/hip_runtime.h>
#include <hip/hip_bf16.h>
#include <cstdint>
#include <cstddef>

// ---------------------------------------------------------------------------
// CDNA5 (gfx1250) ContentEncoder: DCNv2 stack via bf16 WMMA GEMMs.
// GEMM: register-blocked 32Mx64N per wave -> 8 v_wmma_f32_16x16x32_bf16 per
// 32-K step at 0.75KB loaded per WMMA.
// ---------------------------------------------------------------------------

typedef __attribute__((ext_vector_type(16))) __bf16 v16bf;
typedef __attribute__((ext_vector_type(8)))  float  v8f;

struct BFrag { uint4 lo; uint4 hi; };   // 32 bytes == one v16bf A fragment

static __device__ __forceinline__ uint16_t f2bf(float f) {
  uint32_t u = __float_as_uint(f);
  u += 0x7FFFu + ((u >> 16) & 1u);      // round-to-nearest-even
  return (uint16_t)(u >> 16);
}

// ---- pack fp32 weights [Cout, Ktrue] -> bf16 B-matrix [Npad][Kpad] ---------
__global__ void wpack_kernel(const float* __restrict__ w, uint16_t* __restrict__ wm,
                             int Cout, int Ktrue, int Kpad, int Npad) {
  long long idx = (long long)blockIdx.x * blockDim.x + threadIdx.x;
  long long total = (long long)Npad * Kpad;
  if (idx >= total) return;
  int k = (int)(idx % Kpad);
  int n = (int)(idx / Kpad);
  float v = (n < Cout && k < Ktrue) ? w[(size_t)n * Ktrue + k] : 0.0f;
  wm[idx] = f2bf(v);
}

// ---- im2col (zero-pad or reflect-pad) into bf16 columns [M][Kpad] ----------
__global__ void im2col_kernel(const float* __restrict__ x, uint16_t* __restrict__ col,
                              int B, int C, int H, int W, int KH, int KW,
                              int stride, int pad, int Ho, int Wo,
                              int Ktrue, int Kpad, int reflect) {
  long long idx = (long long)blockIdx.x * blockDim.x + threadIdx.x;
  int P = Ho * Wo;
  long long total = (long long)B * P * Kpad;
  if (idx >= total) return;
  int kidx = (int)(idx % Kpad);
  long long m = idx / Kpad;
  if (kidx >= Ktrue) { col[idx] = 0; return; }
  int p = (int)(m % P);
  int b = (int)(m / P);
  int wo = p % Wo, ho = p / Wo;
  int kk = KH * KW;
  int c = kidx / kk, t = kidx % kk;
  int ky = t / KW, kx = t % KW;
  int iy = ho * stride - pad + ky;
  int ix = wo * stride - pad + kx;
  float v;
  if (reflect) {
    iy = (iy < 0) ? -iy : ((iy >= H) ? 2 * H - 2 - iy : iy);
    ix = (ix < 0) ? -ix : ((ix >= W) ? 2 * W - 2 - ix : ix);
    v = x[(((size_t)b * C + c) * H + iy) * W + ix];
  } else {
    bool ok = (iy >= 0) & (iy < H) & (ix >= 0) & (ix < W);
    v = ok ? x[(((size_t)b * C + c) * H + iy) * W + ix] : 0.0f;
  }
  col[idx] = f2bf(v);
}

// ---- DCNv2 sampling: build modulated bilinear columns [M][Kpad] ------------
__global__ void deform_sample_kernel(const float* __restrict__ x, const float* __restrict__ om,
                                     uint16_t* __restrict__ col,
                                     int B, int Cin, int H, int W, int K, int KH, int KW,
                                     int stride, int pad, int Ho, int Wo, int Kpad) {
  long long idx = (long long)blockIdx.x * blockDim.x + threadIdx.x;
  int P = Ho * Wo;
  long long total = (long long)B * P * K;
  if (idx >= total) return;
  int k = (int)(idx % K);
  long long m = idx / K;
  int p = (int)(m % P);
  int b = (int)(m / P);
  int wo = p % Wo, ho = p / Wo;
  int ky = k / KW, kx = k % KW;
  const float* omb = om + (size_t)b * 3 * K * P;
  float oy = omb[(size_t)k * P + p];
  float ox = omb[(size_t)(K + k) * P + p];
  float mr = omb[(size_t)(2 * K + k) * P + p];
  float mask = 1.0f / (1.0f + __expf(-mr));
  float sy = (float)(ho * stride - pad + ky) + oy;
  float sx = (float)(wo * stride - pad + kx) + ox;
  float y0f = floorf(sy), x0f = floorf(sx);
  float dy = sy - y0f, dx = sx - x0f;
  int y0 = (int)y0f, x0 = (int)x0f;
  int y1 = y0 + 1, x1 = x0 + 1;
  float w00 = (1.f - dy) * (1.f - dx) * mask;
  float w01 = (1.f - dy) * dx * mask;
  float w10 = dy * (1.f - dx) * mask;
  float w11 = dy * dx * mask;
  if (!((y0 >= 0) & (y0 < H) & (x0 >= 0) & (x0 < W))) w00 = 0.f;
  if (!((y0 >= 0) & (y0 < H) & (x1 >= 0) & (x1 < W))) w01 = 0.f;
  if (!((y1 >= 0) & (y1 < H) & (x0 >= 0) & (x0 < W))) w10 = 0.f;
  if (!((y1 >= 0) & (y1 < H) & (x1 >= 0) & (x1 < W))) w11 = 0.f;
  int cy0 = min(max(y0, 0), H - 1), cy1 = min(max(y1, 0), H - 1);
  int cx0 = min(max(x0, 0), W - 1), cx1 = min(max(x1, 0), W - 1);
  size_t i00 = (size_t)cy0 * W + cx0, i01 = (size_t)cy0 * W + cx1;
  size_t i10 = (size_t)cy1 * W + cx0, i11 = (size_t)cy1 * W + cx1;
  const float* xb = x + (size_t)b * Cin * H * W;
  uint16_t* cm = col + m * Kpad + k;
  for (int c = 0; c < Cin; ++c) {
    const float* xc = xb + (size_t)c * H * W;
    float v = w00 * xc[i00] + w01 * xc[i01] + w10 * xc[i10] + w11 * xc[i11];
    cm[(size_t)c * K] = f2bf(v);
  }
}

// zero-fill column K-tail [Ktrue, Kpad) (needed when Ktrue not mult of 32)
__global__ void ztail_kernel(uint16_t* __restrict__ col, int Kpad, int Ktrue, long long M) {
  int tail = Kpad - Ktrue;
  long long idx = (long long)blockIdx.x * blockDim.x + threadIdx.x;
  long long total = M * tail;
  if (idx >= total) return;
  long long m = idx / tail;
  int t = (int)(idx % tail);
  col[m * Kpad + Ktrue + t] = 0;
}

// ---- bf16 WMMA GEMM: out[b,n,p] = sum_k A[m,k]*B[n,k] + bias[n] ------------
// A: [M][Kpad] bf16 columns; Bw: [Npad][Kpad] bf16 packed weights.
// Each wave: 32M x 64N macro-tile (2 A frags x 4 B frags -> 8 accumulators).
// Block = 8 waves stacked along M -> 256M x 64N. Requires M % 256 == 0.
__global__ __launch_bounds__(256) void gemm_wmma_kernel(
    const uint16_t* __restrict__ A, const uint16_t* __restrict__ Bw,
    const float* __restrict__ bias, float* __restrict__ out,
    int M, int N, int Kpad, int Cout, int P) {
  int lane = threadIdx.x & 31;
  int wave = threadIdx.x >> 5;              // 0..7 along M
  int m0 = blockIdx.y * 256 + wave * 32;    // wave covers [m0, m0+32)
  int n0 = blockIdx.x * 64;                 // block covers [n0, n0+64)
  if (m0 >= M) return;                      // wave-uniform -> EXEC all-ones in WMMA
  int l15 = lane & 15;
  int hi = lane >> 4;                       // K half select per CDNA5 16-bit layout
  const uint16_t* Ap0 = A + (size_t)(m0 + l15) * Kpad + hi * 8;
  const uint16_t* Ap1 = Ap0 + (size_t)16 * Kpad;
  const uint16_t* Bp  = Bw + (size_t)(n0 + l15) * Kpad + hi * 8;
  const size_t bstep = (size_t)16 * Kpad;
  v8f acc[2][4] = {};
#pragma unroll 2
  for (int kk = 0; kk < Kpad; kk += 32) {
    __builtin_prefetch(Ap0 + kk + 512, 0, 1);       // global_prefetch_b8
    __builtin_prefetch(Ap1 + kk + 512, 0, 1);
    BFrag af0, af1;
    af0.lo = *(const uint4*)(Ap0 + kk);             // lanes<16: K 0..7 | >=16: K 8..15
    af0.hi = *(const uint4*)(Ap0 + kk + 16);        // lanes<16: K16..23| >=16: K24..31
    af1.lo = *(const uint4*)(Ap1 + kk);
    af1.hi = *(const uint4*)(Ap1 + kk + 16);
    v16bf a0 = __builtin_bit_cast(v16bf, af0);
    v16bf a1 = __builtin_bit_cast(v16bf, af1);
#pragma unroll
    for (int nt = 0; nt < 4; ++nt) {
      BFrag bfr;
      bfr.lo = *(const uint4*)(Bp + (size_t)nt * bstep + kk);
      bfr.hi = *(const uint4*)(Bp + (size_t)nt * bstep + kk + 16);
      v16bf bv = __builtin_bit_cast(v16bf, bfr);
      acc[0][nt] = __builtin_amdgcn_wmma_f32_16x16x32_bf16(
          false, a0, false, bv, (short)0, acc[0][nt], false, false);
      acc[1][nt] = __builtin_amdgcn_wmma_f32_16x16x32_bf16(
          false, a1, false, bv, (short)0, acc[1][nt], false, false);
    }
  }
#pragma unroll
  for (int nt = 0; nt < 4; ++nt) {
    int n = n0 + nt * 16 + l15;
    if (n >= N) continue;
    float bb = bias[n];
#pragma unroll
    for (int mt = 0; mt < 2; ++mt) {
      int mbase = m0 + mt * 16 + hi * 8;
#pragma unroll
      for (int rr = 0; rr < 8; ++rr) {
        int m = mbase + rr;
        int b = m / P, p = m - b * P;
        out[((size_t)b * Cout + n) * P + p] = acc[mt][nt][rr] + bb;
      }
    }
  }
}

// ---- instance norm: stats then in-place apply ------------------------------
__global__ __launch_bounds__(256) void in_stats_kernel(const float* __restrict__ x,
                                                       float* __restrict__ stats, int P) {
  int bc = blockIdx.x;
  const float* xp = x + (size_t)bc * P;
  float s = 0.f, s2 = 0.f;
  for (int i = threadIdx.x; i < P; i += 256) { float v = xp[i]; s += v; s2 += v * v; }
  __shared__ float sh[512];
  sh[threadIdx.x] = s; sh[256 + threadIdx.x] = s2;
  __syncthreads();
  for (int o = 128; o > 0; o >>= 1) {
    if ((int)threadIdx.x < o) {
      sh[threadIdx.x] += sh[threadIdx.x + o];
      sh[256 + threadIdx.x] += sh[256 + threadIdx.x + o];
    }
    __syncthreads();
  }
  if (threadIdx.x == 0) {
    float mean = sh[0] / (float)P;
    float var  = sh[256] / (float)P - mean * mean;
    stats[2 * bc] = mean; stats[2 * bc + 1] = var;
  }
}

__global__ void in_apply_kernel(float* __restrict__ x, const float* __restrict__ stats,
                                int P, long long total, int relu) {
  long long idx = (long long)blockIdx.x * blockDim.x + threadIdx.x;
  if (idx >= total) return;
  int bc = (int)(idx / P);
  float mean = stats[2 * bc], var = stats[2 * bc + 1];
  float v = (x[idx] - mean) * rsqrtf(var + 1e-5f);
  if (relu) v = fmaxf(v, 0.f);
  x[idx] = v;
}

__global__ void add_kernel(float* __restrict__ h, const float* __restrict__ y, long long n) {
  long long idx = (long long)blockIdx.x * blockDim.x + threadIdx.x;
  if (idx < n) h[idx] += y[idx];
}

// ---------------------------------------------------------------------------
static inline unsigned cdivu(long long a, long long b) { return (unsigned)((a + b - 1) / b); }

extern "C" void kernel_launch(void* const* d_in, const int* in_sizes, int n_in,
                              void* d_out, int out_size, void* d_ws, size_t ws_size,
                              hipStream_t stream) {
  (void)in_sizes; (void)n_in; (void)out_size; (void)ws_size;
  const float* x      = (const float*)d_in[0];
  const float* w_off1 = (const float*)d_in[1];  const float* b_off1 = (const float*)d_in[2];
  const float* w1     = (const float*)d_in[3];  const float* b1     = (const float*)d_in[4];
  const float* w_off2 = (const float*)d_in[5];  const float* b_off2 = (const float*)d_in[6];
  const float* w2     = (const float*)d_in[7];  const float* b2     = (const float*)d_in[8];
  const float* w_off3 = (const float*)d_in[9];  const float* b_off3 = (const float*)d_in[10];
  const float* w3     = (const float*)d_in[11]; const float* b3     = (const float*)d_in[12];

  float* out_h     = (float*)d_out;                    // [16,256,32,32]
  float* out_skip2 = out_h + (size_t)4194304;          // [16,128,64,64]
  float* out_skip3 = out_h + (size_t)12582912;         // [16,256,32,32]

  // workspace layout
  char* ws = (char*)d_ws;
  uint16_t* colA = (uint16_t*)ws;                              // 134,217,728 B (max col)
  float*    om   = (float*)(ws + 134217728ull);                // 154,140,672 B (max 3K maps)
  float*    h1   = (float*)(ws + 134217728ull + 154140672ull); // 67,108,864 B
  float*    ybuf = h1;                                         // reuse: h1 dead before residuals
  uint16_t* wmat = (uint16_t*)(ws + 134217728ull + 154140672ull + 67108864ull); // 2 MB
  float*    stats = (float*)(ws + 134217728ull + 154140672ull + 67108864ull + 2097152ull);

  const int TPB = 256;
  auto gemm = [&](const uint16_t* A, const uint16_t* Bw, const float* bias, float* out,
                  int M, int N, int Kpad, int Npad, int P) {
    dim3 grid(Npad / 64, (unsigned)(M / 256));
    gemm_wmma_kernel<<<grid, TPB, 0, stream>>>(A, Bw, bias, out, M, N, Kpad, N, P);
  };

  // ============== Layer 1: deform conv 7x7 s1 p3, 3 -> 64, 128x128 ==========
  { // offset conv: N=147, Ktrue=147, Kpad=160, M=262144, P=16384
    wpack_kernel<<<cdivu(192ll * 160, TPB), TPB, 0, stream>>>(w_off1, wmat, 147, 147, 160, 192);
    im2col_kernel<<<cdivu(262144ll * 160, TPB), TPB, 0, stream>>>(x, colA,
        16, 3, 128, 128, 7, 7, 1, 3, 128, 128, 147, 160, 0);
    gemm(colA, wmat, b_off1, om, 262144, 147, 160, 192, 16384);
    // modulated sampling -> columns
    wpack_kernel<<<cdivu(64ll * 160, TPB), TPB, 0, stream>>>(w1, wmat, 64, 147, 160, 64);
    deform_sample_kernel<<<cdivu(262144ll * 49, TPB), TPB, 0, stream>>>(x, om, colA,
        16, 3, 128, 128, 49, 7, 7, 1, 3, 128, 128, 160);
    ztail_kernel<<<cdivu(262144ll * 13, TPB), TPB, 0, stream>>>(colA, 160, 147, 262144);
    gemm(colA, wmat, b1, h1, 262144, 64, 160, 64, 16384);
    in_stats_kernel<<<16 * 64, TPB, 0, stream>>>(h1, stats, 16384);
    in_apply_kernel<<<cdivu(16ll * 64 * 16384, TPB), TPB, 0, stream>>>(
        h1, stats, 16384, 16ll * 64 * 16384, 1);
  }

  // ============== Layer 2: deform conv 4x4 s2 p1, 64 -> 128, out 64x64 ======
  { // offset conv: N=48, K=1024, M=65536, P=4096
    wpack_kernel<<<cdivu(64ll * 1024, TPB), TPB, 0, stream>>>(w_off2, wmat, 48, 1024, 1024, 64);
    im2col_kernel<<<cdivu(65536ll * 1024, TPB), TPB, 0, stream>>>(h1, colA,
        16, 64, 128, 128, 4, 4, 2, 1, 64, 64, 1024, 1024, 0);
    gemm(colA, wmat, b_off2, om, 65536, 48, 1024, 64, 4096);
    wpack_kernel<<<cdivu(128ll * 1024, TPB), TPB, 0, stream>>>(w2, wmat, 128, 1024, 1024, 128);
    deform_sample_kernel<<<cdivu(65536ll * 16, TPB), TPB, 0, stream>>>(h1, om, colA,
        16, 64, 128, 128, 16, 4, 4, 2, 1, 64, 64, 1024);
    gemm(colA, wmat, b2, out_skip2, 65536, 128, 1024, 128, 4096);
    in_stats_kernel<<<16 * 128, TPB, 0, stream>>>(out_skip2, stats, 4096);
    in_apply_kernel<<<cdivu(16ll * 128 * 4096, TPB), TPB, 0, stream>>>(
        out_skip2, stats, 4096, 16ll * 128 * 4096, 1);  // skip2 done (in d_out)
  }

  // ============== Layer 3: deform conv 4x4 s2 p1, 128 -> 256, out 32x32 =====
  { // offset conv: N=48, K=2048, M=16384, P=1024
    wpack_kernel<<<cdivu(64ll * 2048, TPB), TPB, 0, stream>>>(w_off3, wmat, 48, 2048, 2048, 64);
    im2col_kernel<<<cdivu(16384ll * 2048, TPB), TPB, 0, stream>>>(out_skip2, colA,
        16, 128, 64, 64, 4, 4, 2, 1, 32, 32, 2048, 2048, 0);
    gemm(colA, wmat, b_off3, om, 16384, 48, 2048, 64, 1024);
    wpack_kernel<<<cdivu(256ll * 2048, TPB), TPB, 0, stream>>>(w3, wmat, 256, 2048, 2048, 256);
    deform_sample_kernel<<<cdivu(16384ll * 16, TPB), TPB, 0, stream>>>(out_skip2, om, colA,
        16, 128, 64, 64, 16, 4, 4, 2, 1, 32, 32, 2048);
    gemm(colA, wmat, b3, out_skip3, 16384, 256, 2048, 256, 1024);
    in_stats_kernel<<<16 * 256, TPB, 0, stream>>>(out_skip3, stats, 1024);
    in_apply_kernel<<<cdivu(16ll * 256 * 1024, TPB), TPB, 0, stream>>>(
        out_skip3, stats, 1024, 16ll * 256 * 1024, 1);  // skip3 done (in d_out)
  }

  // h starts as skip3
  hipMemcpyAsync(out_h, out_skip3, (size_t)4194304 * sizeof(float),
                 hipMemcpyDeviceToDevice, stream);

  // ============== 2 residual blocks: reflect 3x3, 256ch, 32x32 ==============
  for (int r = 0; r < 2; ++r) {
    const float* wa = (const float*)d_in[13 + 4 * r];
    const float* ba = (const float*)d_in[14 + 4 * r];
    const float* wb = (const float*)d_in[15 + 4 * r];
    const float* bb = (const float*)d_in[16 + 4 * r];
    // y = relu(IN(conv3_reflect(h, wa, ba)))
    wpack_kernel<<<cdivu(256ll * 2304, TPB), TPB, 0, stream>>>(wa, wmat, 256, 2304, 2304, 256);
    im2col_kernel<<<cdivu(16384ll * 2304, TPB), TPB, 0, stream>>>(out_h, colA,
        16, 256, 32, 32, 3, 3, 1, 1, 32, 32, 2304, 2304, 1);
    gemm(colA, wmat, ba, ybuf, 16384, 256, 2304, 256, 1024);
    in_stats_kernel<<<16 * 256, TPB, 0, stream>>>(ybuf, stats, 1024);
    in_apply_kernel<<<cdivu(4194304ll, TPB), TPB, 0, stream>>>(
        ybuf, stats, 1024, 4194304ll, 1);
    // y = IN(conv3_reflect(y, wb, bb));  h += y
    wpack_kernel<<<cdivu(256ll * 2304, TPB), TPB, 0, stream>>>(wb, wmat, 256, 2304, 2304, 256);
    im2col_kernel<<<cdivu(16384ll * 2304, TPB), TPB, 0, stream>>>(ybuf, colA,
        16, 256, 32, 32, 3, 3, 1, 1, 32, 32, 2304, 2304, 1);
    gemm(colA, wmat, bb, ybuf, 16384, 256, 2304, 256, 1024);   // colA fully built; in-place OK
    in_stats_kernel<<<16 * 256, TPB, 0, stream>>>(ybuf, stats, 1024);
    in_apply_kernel<<<cdivu(4194304ll, TPB), TPB, 0, stream>>>(
        ybuf, stats, 1024, 4194304ll, 0);
    add_kernel<<<cdivu(4194304ll, TPB), TPB, 0, stream>>>(out_h, ybuf, 4194304ll);
  }
}
